// DynamicGraphConvolution_37735582662907
// MI455X (gfx1250) — matrix operand order repaired and checked by reading, not compiled, not run on hardware
//
#include <hip/hip_runtime.h>
#include <hip/hip_bf16.h>

typedef __attribute__((ext_vector_type(16))) __bf16 v16bf;
typedef __attribute__((ext_vector_type(8)))  __bf16 bf8;
typedef __attribute__((ext_vector_type(8)))  float  v8f;

#define EPSBN 1e-5f
#define SLOPE 0.2f

__device__ __forceinline__ __bf16 f2bf(float f) {
  unsigned u = __float_as_uint(f);
  u += 0x7FFFu + ((u >> 16) & 1u);
  unsigned short h = (unsigned short)(u >> 16);
  return __builtin_bit_cast(__bf16, h);
}
__device__ __forceinline__ float lky(float y) { return y >= 0.f ? y : SLOPE * y; }

// ---------------------------------------------------------------------------
// bf16 WMMA GEMM: C[M,N] f32 = A[M,K] * Bt[N,K]^T   (B given pre-transposed,
// both operands contiguous in K). Batched over blockIdx.z.
// Block tile 128x64, 8 waves; wave w -> rows [16w,16w+16) x all 64 cols
// (1 A-frag reused by 4 WMMAs per 32-deep K step).
// Double-buffered LDS + register staging: one barrier per K step.
// ---------------------------------------------------------------------------
__global__ __launch_bounds__(256) void k_gemm_bf16(
    const __bf16* __restrict__ Ag, const __bf16* __restrict__ Btg, float* __restrict__ Cg,
    int M, int N, int K, int lda, int ldbt, int ldc,
    long long sA, long long sB, long long sC)
{
  __shared__ __align__(16) __bf16 As[2][128][32];   // [buf][m][k]
  __shared__ __align__(16) __bf16 Bs[2][64][32];    // [buf][n][k]

  const __bf16* A  = Ag  + (long long)blockIdx.z * sA;
  const __bf16* Bt = Btg + (long long)blockIdx.z * sB;
  float*        C  = Cg  + (long long)blockIdx.z * sC;

  const int tid  = threadIdx.x;
  const int wave = tid >> 5, lane = tid & 31;
  const int bm = blockIdx.y * 128, bn = blockIdx.x * 64;
  const int wr = wave * 16;
  const int lm = lane & 15, lh = lane >> 4;

  // tile-fill mapping: 16B (8 x bf16) chunks
  const int a0r = tid >> 2;            // rows 0..63
  const int a1r = a0r + 64;            // rows 64..127
  const int kc  = (tid & 3) * 8;
  const int gm0 = bm + a0r, gm1 = bm + a1r, gnl = bn + a0r;

  v8f acc0 = {}, acc1 = {}, acc2 = {}, acc3 = {};
  bf8 rA0 = {}, rA1 = {}, rB = {};

  // prologue: load k0=0 into regs, store to buffer 0
  if (gm0 < M) rA0 = *reinterpret_cast<const bf8*>(A  + (long long)gm0 * lda  + kc);
  if (gm1 < M) rA1 = *reinterpret_cast<const bf8*>(A  + (long long)gm1 * lda  + kc);
  if (gnl < N) rB  = *reinterpret_cast<const bf8*>(Bt + (long long)gnl * ldbt + kc);
  *reinterpret_cast<bf8*>(&As[0][a0r][kc]) = rA0;
  *reinterpret_cast<bf8*>(&As[0][a1r][kc]) = rA1;
  *reinterpret_cast<bf8*>(&Bs[0][a0r][kc]) = rB;

  int buf = 0;
  for (int k0 = 0; k0 < K; k0 += 32) {
    __syncthreads();
    const bool more = (k0 + 32) < K;
    if (more) { // stage next tile in registers (overlaps with WMMAs below)
      rA0 = bf8{}; rA1 = bf8{}; rB = bf8{};
      if (gm0 < M) rA0 = *reinterpret_cast<const bf8*>(A  + (long long)gm0 * lda  + k0 + 32 + kc);
      if (gm1 < M) rA1 = *reinterpret_cast<const bf8*>(A  + (long long)gm1 * lda  + k0 + 32 + kc);
      if (gnl < N) rB  = *reinterpret_cast<const bf8*>(Bt + (long long)gnl * ldbt + k0 + 32 + kc);
    }

    // A fragment (16-bit A 16x32): lane lm=M, half lh:
    //   elem j = K[16*lh + j] (same k order as B)
    const bf8* arow = reinterpret_cast<const bf8*>(&As[buf][wr + lm][0]);
    bf8 alo = arow[2 * lh], ahi = arow[2 * lh + 1];
    v16bf a = __builtin_shufflevector(alo, ahi, 0,1,2,3,4,5,6,7,8,9,10,11,12,13,14,15);

    // B fragments (B 32x16 layout): lane lm=N, half lh: elem j = K[16*lh + j]
    const bf8* br0 = reinterpret_cast<const bf8*>(&Bs[buf][ 0 + lm][0]);
    const bf8* br1 = reinterpret_cast<const bf8*>(&Bs[buf][16 + lm][0]);
    const bf8* br2 = reinterpret_cast<const bf8*>(&Bs[buf][32 + lm][0]);
    const bf8* br3 = reinterpret_cast<const bf8*>(&Bs[buf][48 + lm][0]);
    v16bf b0 = __builtin_shufflevector(br0[2*lh], br0[2*lh+1], 0,1,2,3,4,5,6,7,8,9,10,11,12,13,14,15);
    v16bf b1 = __builtin_shufflevector(br1[2*lh], br1[2*lh+1], 0,1,2,3,4,5,6,7,8,9,10,11,12,13,14,15);
    v16bf b2 = __builtin_shufflevector(br2[2*lh], br2[2*lh+1], 0,1,2,3,4,5,6,7,8,9,10,11,12,13,14,15);
    v16bf b3 = __builtin_shufflevector(br3[2*lh], br3[2*lh+1], 0,1,2,3,4,5,6,7,8,9,10,11,12,13,14,15);

    acc0 = __builtin_amdgcn_wmma_f32_16x16x32_bf16(false, a, false, b0, (short)0, acc0, false, false);
    acc1 = __builtin_amdgcn_wmma_f32_16x16x32_bf16(false, a, false, b1, (short)0, acc1, false, false);
    acc2 = __builtin_amdgcn_wmma_f32_16x16x32_bf16(false, a, false, b2, (short)0, acc2, false, false);
    acc3 = __builtin_amdgcn_wmma_f32_16x16x32_bf16(false, a, false, b3, (short)0, acc3, false, false);

    if (more) { // fill the other buffer for next iteration
      int nb = buf ^ 1;
      *reinterpret_cast<bf8*>(&As[nb][a0r][kc]) = rA0;
      *reinterpret_cast<bf8*>(&As[nb][a1r][kc]) = rA1;
      *reinterpret_cast<bf8*>(&Bs[nb][a0r][kc]) = rB;
    }
    buf ^= 1;
  }

  // D layout: VGPR r -> M = r + 8*lh ; N = lane%16
  #pragma unroll
  for (int r = 0; r < 8; ++r) {
    int gm = bm + wr + r + lh * 8;
    if (gm < M) {
      float* Crow = C + (long long)gm * ldc;
      int gn = bn + lm;
      if (gn      < N) Crow[gn]      = acc0[r];
      if (gn + 16 < N) Crow[gn + 16] = acc1[r];
      if (gn + 32 < N) Crow[gn + 32] = acc2[r];
      if (gn + 48 < N) Crow[gn + 48] = acc3[r];
    }
  }
}

// ---------------------------------------------------------------------------
// adj preprocessing: global minmax of adj_param, sadj, D=rsqrt(rowsum),
// adjn[n,m] = D[n]*sadj[m,n]*D[m]
// ---------------------------------------------------------------------------
__global__ void k_adj_prep(const float* __restrict__ ap, float* __restrict__ sadj,
                           float* __restrict__ adjn)
{
  const int NN = 6400;
  __shared__ float smin[256], smax[256], sD[80];
  __shared__ float s_mn, s_sc;
  int t = threadIdx.x;
  float mn = 3.4e38f, mx = -3.4e38f;
  for (int i = t; i < NN; i += 256) { float v = ap[i]; mn = fminf(mn, v); mx = fmaxf(mx, v); }
  smin[t] = mn; smax[t] = mx; __syncthreads();
  for (int s = 128; s > 0; s >>= 1) {
    if (t < s) { smin[t] = fminf(smin[t], smin[t+s]); smax[t] = fmaxf(smax[t], smax[t+s]); }
    __syncthreads();
  }
  if (t == 0) { s_mn = smin[0]; s_sc = 1.0f / (smax[0] - smin[0]); }
  __syncthreads();
  float m0 = s_mn, sc = s_sc;
  for (int i = t; i < NN; i += 256) sadj[i] = (ap[i] - m0) * sc;
  if (t < 80) {
    float s = 0.f;
    for (int m = 0; m < 80; ++m) s += (ap[t * 80 + m] - m0) * sc;
    sD[t] = rsqrtf(s);
  }
  __syncthreads();
  for (int i = t; i < NN; i += 256) {
    int n = i / 80, m = i % 80;
    adjn[i] = sD[n] * ((ap[m * 80 + n] - m0) * sc) * sD[m];
  }
}

// fp32 -> bf16 convert, optional transpose
__global__ void k_cvt_bf16(const float* __restrict__ in, __bf16* __restrict__ out,
                           int rows, int cols, int ldi, int ldo, int trans)
{
  long long idx = (long long)blockIdx.x * 256 + threadIdx.x;
  long long total = (long long)rows * cols;
  if (idx >= total) return;
  int r = (int)(idx / cols), c = (int)(idx % cols);
  float v = in[(long long)r * ldi + c];
  if (trans) out[(long long)c * ldo + r] = f2bf(v);
  else       out[(long long)r * ldo + c] = f2bf(v);
}

// x (B,C,N) f32 -> xT[(b*N+n)*C + c] bf16
__global__ void k_pack_xT(const float* __restrict__ x, __bf16* __restrict__ xT,
                          int Bb, int C, int Nn)
{
  long long idx = (long long)blockIdx.x * 256 + threadIdx.x;
  long long total = (long long)Bb * C * Nn;
  if (idx >= total) return;
  int n = (int)(idx % Nn); long long t = idx / Nn;
  int c = (int)(t % C);    int b = (int)(t / C);
  xT[((long long)b * Nn + n) * C + c] = f2bf(x[idx]);
}

// H[(b*80+n)*1024+o] = sum_m adjn[n,m] * S[(b*80+m)*1024+o]
__global__ __launch_bounds__(256) void k_graph_static(const float* __restrict__ adjn,
    const float* __restrict__ S, float* __restrict__ H)
{
  __shared__ float co[80];
  int o = blockIdx.x * 256 + threadIdx.x;
  int n = blockIdx.y, b = blockIdx.z;
  if (threadIdx.x < 80) co[threadIdx.x] = adjn[n * 80 + threadIdx.x];
  __syncthreads();
  const float* Sb = S + (long long)b * 80 * 1024;
  float acc = 0.f;
  #pragma unroll 4
  for (int m = 0; m < 80; ++m) acc += co[m] * Sb[m * 1024 + o];
  H[((long long)b * 80 + n) * 1024 + o] = acc;
}

// same but tadj[b,n,m] = Dd[b,n]*dadj[b,m,n]*Dd[b,m]
__global__ __launch_bounds__(256) void k_graph_dyn(const float* __restrict__ dadj,
    const float* __restrict__ Dd, const float* __restrict__ S, float* __restrict__ H)
{
  __shared__ float co[80];
  int o = blockIdx.x * 256 + threadIdx.x;
  int n = blockIdx.y, b = blockIdx.z;
  if (threadIdx.x < 80) {
    int m = threadIdx.x;
    co[m] = Dd[b * 80 + n] * dadj[((long long)b * 80 + m) * 80 + n] * Dd[b * 80 + m];
  }
  __syncthreads();
  const float* Sb = S + (long long)b * 80 * 1024;
  float acc = 0.f;
  #pragma unroll 4
  for (int m = 0; m < 80; ++m) acc += co[m] * Sb[m * 1024 + o];
  H[((long long)b * 80 + n) * 1024 + o] = acc;
}

// deterministic column sum/sumsq partials over row chunks
__global__ void k_colsum_partial(const float* __restrict__ X, int R, int C, int rowsPer,
                                 float* __restrict__ ps, float* __restrict__ pq)
{
  int c = blockIdx.x * 256 + threadIdx.x;
  if (c >= C) return;
  int chunk = blockIdx.y;
  int r0 = chunk * rowsPer;
  int r1 = r0 + rowsPer; if (r1 > R) r1 = R;
  float s = 0.f, q = 0.f;
  for (int r = r0; r < r1; ++r) { float v = X[(long long)r * C + c]; s += v; q += v * v; }
  ps[(long long)chunk * C + c] = s;
  pq[(long long)chunk * C + c] = q;
}

__global__ void k_colsum_final(const float* __restrict__ ps, const float* __restrict__ pq,
                               int C, int chunks, float invR,
                               float* __restrict__ mu, float* __restrict__ var)
{
  int c = blockIdx.x * 256 + threadIdx.x;
  if (c >= C) return;
  float s = 0.f, q = 0.f;
  for (int k = 0; k < chunks; ++k) { s += ps[(long long)k * C + c]; q += pq[(long long)k * C + c]; }
  float m = s * invR;
  mu[c] = m; var[c] = q * invR - m * m;
}

// x2[b,c,n] = x[b,c,n] + leaky(bn(H[(b*N+n)*C+c]))
__global__ void k_bn_leaky_add_x2(const float* __restrict__ x, const float* __restrict__ H,
    const float* __restrict__ mu, const float* __restrict__ var,
    const float* __restrict__ g, const float* __restrict__ be,
    float* __restrict__ x2, int Bb, int C, int Nn)
{
  long long idx = (long long)blockIdx.x * 256 + threadIdx.x;
  long long total = (long long)Bb * Nn * C;
  if (idx >= total) return;
  int c = (int)(idx % C); long long row = idx / C;
  int n = (int)(row % Nn); int b = (int)(row / Nn);
  float v = H[idx];
  float y = g[c] * (v - mu[c]) * rsqrtf(var[c] + EPSBN) + be[c];
  y = lky(y);
  long long xo = ((long long)b * C + c) * Nn + n;
  x2[xo] = x[xo] + y;
}

// glb[b,c] = mean_n x2[b,c,n]
__global__ void k_gap(const float* __restrict__ x2, float* __restrict__ glb, int Nn)
{
  int idx = blockIdx.x * 256 + threadIdx.x;
  const float* p = x2 + (long long)idx * Nn;
  float s = 0.f;
  for (int n = 0; n < Nn; ++n) s += p[n];
  glb[idx] = s * (1.0f / Nn);
}

__global__ void k_add_bias(float* __restrict__ y, const float* __restrict__ b, int C, int total)
{
  int idx = blockIdx.x * 256 + threadIdx.x;
  if (idx >= total) return;
  y[idx] += b[idx % C];
}

// glbA_bf[b,c] = bf16(leaky(bn_g(glb2)))
__global__ void k_bn_leaky_pack(const float* __restrict__ X,
    const float* __restrict__ mu, const float* __restrict__ var,
    const float* __restrict__ g, const float* __restrict__ be,
    __bf16* __restrict__ out, int C)
{
  int idx = blockIdx.x * 256 + threadIdx.x;
  int c = idx % C;
  float y = g[c] * (X[idx] - mu[c]) * rsqrtf(var[c] + EPSBN) + be[c];
  out[idx] = f2bf(lky(y));
}

// dadj += term1[b,o] + bias[o]; per-block min/max partials
__global__ void k_dadj_assemble(float* __restrict__ dadj, const float* __restrict__ t1,
    const float* __restrict__ bias, float* __restrict__ pmin, float* __restrict__ pmax)
{
  __shared__ float smin[256], smax[256];
  int idx = blockIdx.x * 256 + threadIdx.x;         // 256*80*80 / 256 = 6400 blocks exactly
  int t = idx / 80; int o = t % 80; int b = t / 80;
  float v = dadj[idx] + t1[b * 80 + o] + bias[o];
  dadj[idx] = v;
  int tt = threadIdx.x;
  smin[tt] = v; smax[tt] = v; __syncthreads();
  for (int s = 128; s > 0; s >>= 1) {
    if (tt < s) { smin[tt] = fminf(smin[tt], smin[tt+s]); smax[tt] = fmaxf(smax[tt], smax[tt+s]); }
    __syncthreads();
  }
  if (tt == 0) { pmin[blockIdx.x] = smin[0]; pmax[blockIdx.x] = smax[0]; }
}

__global__ void k_minmax_final(const float* __restrict__ pmin, const float* __restrict__ pmax,
                               int n, float* __restrict__ mm)
{
  __shared__ float smin[256], smax[256];
  int t = threadIdx.x;
  float mn = 3.4e38f, mx = -3.4e38f;
  for (int i = t; i < n; i += 256) { mn = fminf(mn, pmin[i]); mx = fmaxf(mx, pmax[i]); }
  smin[t] = mn; smax[t] = mx; __syncthreads();
  for (int s = 128; s > 0; s >>= 1) {
    if (t < s) { smin[t] = fminf(smin[t], smin[t+s]); smax[t] = fmaxf(smax[t], smax[t+s]); }
    __syncthreads();
  }
  if (t == 0) { mm[0] = smin[0]; mm[1] = 1.0f / (smax[0] - smin[0]); }
}

__global__ void k_dadj_norm(float* __restrict__ dadj, const float* __restrict__ mm)
{
  int idx = blockIdx.x * 256 + threadIdx.x;
  dadj[idx] = (dadj[idx] - mm[0]) * mm[1];
}

__global__ void k_dd(const float* __restrict__ dadj, float* __restrict__ Dd, int Nn)
{
  int idx = blockIdx.x * 256 + threadIdx.x;
  const float* p = dadj + (long long)idx * Nn;
  float s = 0.f;
  for (int n = 0; n < Nn; ++n) s += p[n];
  Dd[idx] = rsqrtf(s);
}

// per-batch ||out1 - out1@dadj/N||
__global__ void k_loss1(const float* __restrict__ out1, const float* __restrict__ dadj,
                        float* __restrict__ lossA)
{
  __shared__ float sacc[128];
  int b = blockIdx.x, t = threadIdx.x;
  float sq = 0.f;
  if (t < 80) {
    float t1 = 0.f;
    for (int n = 0; n < 80; ++n)
      t1 += out1[b * 80 + n] * dadj[((long long)b * 80 + n) * 80 + t];
    t1 *= (1.0f / 80.0f);
    float d = out1[b * 80 + t] - t1;
    sq = d * d;
  }
  sacc[t] = sq; __syncthreads();
  for (int s = 64; s > 0; s >>= 1) { if (t < s) sacc[t] += sacc[t+s]; __syncthreads(); }
  if (t == 0) lossA[b] = sqrtf(sacc[0]);
}

// per-batch ||dadj[b]-sadj||_F
__global__ void k_loss2(const float* __restrict__ dadj, const float* __restrict__ sadj,
                        float* __restrict__ lossB)
{
  __shared__ float sacc[256];
  int b = blockIdx.x, t = threadIdx.x;
  const float* p = dadj + (long long)b * 6400;
  float sq = 0.f;
  for (int i = t; i < 6400; i += 256) { float d = p[i] - sadj[i]; sq += d * d; }
  sacc[t] = sq; __syncthreads();
  for (int s = 128; s > 0; s >>= 1) { if (t < s) sacc[t] += sacc[t+s]; __syncthreads(); }
  if (t == 0) lossB[b] = sqrtf(sacc[0]);
}

__global__ void k_loss_final(const float* __restrict__ la, const float* __restrict__ lb,
                             int n, float* __restrict__ out)
{
  __shared__ float s[256];
  int t = threadIdx.x;
  float a = 0.f;
  for (int i = t; i < n; i += 256) a += la[i] + lb[i];
  s[t] = a; __syncthreads();
  for (int k = 128; k > 0; k >>= 1) { if (t < k) s[t] += s[t+k]; __syncthreads(); }
  if (t == 0) out[0] = s[0];
}

// out[b,c,n] = leaky(bn(H[(b*N+n)*C+c]))  (fused final transpose)
__global__ void k_bn_leaky_out(const float* __restrict__ H,
    const float* __restrict__ mu, const float* __restrict__ var,
    const float* __restrict__ g, const float* __restrict__ be,
    float* __restrict__ out, int Bb, int C, int Nn)
{
  long long idx = (long long)blockIdx.x * 256 + threadIdx.x;
  long long total = (long long)Bb * Nn * C;
  if (idx >= total) return;
  int c = (int)(idx % C); long long row = idx / C;
  int n = (int)(row % Nn); int b = (int)(row / Nn);
  float v = H[idx];
  float y = g[c] * (v - mu[c]) * rsqrtf(var[c] + EPSBN) + be[c];
  out[((long long)b * C + c) * Nn + n] = lky(y);
}

// ---------------------------------------------------------------------------
extern "C" void kernel_launch(void* const* d_in, const int* in_sizes, int n_in,
                              void* d_out, int out_size, void* d_ws, size_t ws_size,
                              hipStream_t stream)
{
  (void)in_sizes; (void)n_in; (void)out_size; (void)ws_size;
  const int B = 256, C = 1024, Nn = 80;
  const long long RN = (long long)B * Nn;         // 20480
  const long long tX = (long long)B * C * Nn;     // 20971520

  const float* x    = (const float*)d_in[0];
  const float* out1 = (const float*)d_in[1];
  const float* adjp = (const float*)d_in[2];
  const float* Wst  = (const float*)d_in[3];
  const float* Wdy  = (const float*)d_in[4];
  const float* Wg   = (const float*)d_in[5];
  const float* bg   = (const float*)d_in[6];
  const float* Wco  = (const float*)d_in[7];
  const float* bco  = (const float*)d_in[8];
  const float* g1   = (const float*)d_in[9];
  const float* b1   = (const float*)d_in[10];
  const float* gg   = (const float*)d_in[11];
  const float* bb   = (const float*)d_in[12];

  char* w = (char*)d_ws;
  size_t off = 0;
  auto alloc = [&](size_t bytes) -> void* {
    off = (off + 255) & ~(size_t)255;
    void* p = (void*)(w + off); off += bytes; return p;
  };

  float* sadj  = (float*)alloc(6400 * 4);
  float* adjn  = (float*)alloc(6400 * 4);
  float* mu1   = (float*)alloc(C * 4);
  float* var1  = (float*)alloc(C * 4);
  float* mug   = (float*)alloc(C * 4);
  float* varg  = (float*)alloc(C * 4);
  float* mnmx  = (float*)alloc(16 * 4);
  float* ps    = (float*)alloc((size_t)80 * C * 4);
  float* pq    = (float*)alloc((size_t)80 * C * 4);
  float* pmin  = (float*)alloc(6400 * 4);
  float* pmax  = (float*)alloc(6400 * 4);
  float* term1 = (float*)alloc((size_t)B * Nn * 4);
  float* Dd    = (float*)alloc((size_t)B * Nn * 4);
  float* lossA = (float*)alloc(B * 4);
  float* lossB = (float*)alloc(B * 4);
  float* glb   = (float*)alloc((size_t)B * C * 4);
  float* glb2  = (float*)alloc((size_t)B * C * 4);
  __bf16* glbbf= (__bf16*)alloc((size_t)B * C * 2);
  __bf16* WstT = (__bf16*)alloc((size_t)C * C * 2);   // Bt for support GEMM: [o][c]
  __bf16* WdyT = (__bf16*)alloc((size_t)C * C * 2);   // Bt for sup GEMM: [o][c]
  __bf16* Wgb  = (__bf16*)alloc((size_t)C * C * 2);   // Bt = Wg[o][c] (straight)
  __bf16* Wc1  = (__bf16*)alloc((size_t)Nn * C * 2);  // Bt = Wco[o][0:C] (straight)
  __bf16* Wc2  = (__bf16*)alloc((size_t)Nn * C * 2);  // A  = Wco[o][C:2C] (straight)
  float* dadj  = (float*)alloc((size_t)B * Nn * Nn * 4);
  __bf16* xTb  = (__bf16*)alloc((size_t)RN * C * 2);  // xT, later x2T (A and Bt roles)
  float* bufA  = (float*)alloc((size_t)RN * C * 4);
  float* bufB  = (float*)alloc((size_t)RN * C * 4);
  float* x2    = (float*)alloc((size_t)tX * 4);

  float* fout    = (float*)d_out;
  float* lossOut = fout + (size_t)B * C * Nn;

  dim3 blk(256);
  const long long tW = (long long)C * C, tWc = (long long)Nn * C;

  // ---- static path ----
  k_adj_prep<<<1, blk, 0, stream>>>(adjp, sadj, adjn);
  k_cvt_bf16<<<(int)((tW + 255) / 256), blk, 0, stream>>>(Wst, WstT, C, C, C, C, 1);
  k_cvt_bf16<<<(int)((tW + 255) / 256), blk, 0, stream>>>(Wdy, WdyT, C, C, C, C, 1);
  k_cvt_bf16<<<(int)((tW + 255) / 256), blk, 0, stream>>>(Wg, Wgb, C, C, C, C, 0);
  k_cvt_bf16<<<(int)((tWc + 255) / 256), blk, 0, stream>>>(Wco, Wc1, Nn, C, 2 * C, C, 0);
  k_cvt_bf16<<<(int)((tWc + 255) / 256), blk, 0, stream>>>(Wco + C, Wc2, Nn, C, 2 * C, C, 0);
  k_pack_xT<<<(int)((tX + 255) / 256), blk, 0, stream>>>(x, xTb, B, C, Nn);

  // support = xT @ Wst : (20480x1024)x(1024x1024)
  k_gemm_bf16<<<dim3(C / 64, (int)(RN / 128), 1), blk, 0, stream>>>(
      xTb, WstT, bufA, (int)RN, C, C, C, C, C, 0, 0, 0);
  k_graph_static<<<dim3(C / 256, Nn, B), blk, 0, stream>>>(adjn, bufA, bufB);
  k_colsum_partial<<<dim3(C / 256, 80), blk, 0, stream>>>(bufB, (int)RN, C, 256, ps, pq);
  k_colsum_final<<<C / 256, blk, 0, stream>>>(ps, pq, C, 80, 1.0f / (float)RN, mu1, var1);
  k_bn_leaky_add_x2<<<(int)((tX + 255) / 256), blk, 0, stream>>>(x, bufB, mu1, var1, g1, b1, x2, B, C, Nn);

  // ---- dynamic graph construction ----
  k_gap<<<(B * C) / 256, blk, 0, stream>>>(x2, glb, Nn);
  k_cvt_bf16<<<(B * C + 255) / 256, blk, 0, stream>>>(glb, glbbf, B, C, C, C, 0);
  k_gemm_bf16<<<dim3(C / 64, (B + 127) / 128, 1), blk, 0, stream>>>(
      glbbf, Wgb, glb2, B, C, C, C, C, C, 0, 0, 0);
  k_add_bias<<<(B * C) / 256, blk, 0, stream>>>(glb2, bg, C, B * C);
  k_colsum_partial<<<dim3(C / 256, 8), blk, 0, stream>>>(glb2, B, C, 32, ps, pq);
  k_colsum_final<<<C / 256, blk, 0, stream>>>(ps, pq, C, 8, 1.0f / (float)B, mug, varg);
  k_bn_leaky_pack<<<(B * C) / 256, blk, 0, stream>>>(glb2, mug, varg, gg, bb, glbbf, C);

  // x2T (bf16): A for sup GEMM, Bt for term2 GEMM
  k_pack_xT<<<(int)((tX + 255) / 256), blk, 0, stream>>>(x2, xTb, B, C, Nn);

  // term1[b,o] = glbA @ Wco[:, :C]^T
  k_gemm_bf16<<<dim3((Nn + 63) / 64, (B + 127) / 128, 1), blk, 0, stream>>>(
      glbbf, Wc1, term1, B, Nn, C, C, C, Nn, 0, 0, 0);
  // term2[b,o,n] = Wco2 @ x2[b]  (batched; Bt = x2T[b])
  k_gemm_bf16<<<dim3(2, 1, B), blk, 0, stream>>>(
      Wc2, xTb, dadj, Nn, Nn, C, C, C, Nn, 0LL, (long long)Nn * C, (long long)Nn * Nn);

  k_dadj_assemble<<<6400, blk, 0, stream>>>(dadj, term1, bco, pmin, pmax);
  k_minmax_final<<<1, blk, 0, stream>>>(pmin, pmax, 6400, mnmx);
  k_dadj_norm<<<6400, blk, 0, stream>>>(dadj, mnmx);
  k_dd<<<(B * Nn) / 256, blk, 0, stream>>>(dadj, Dd, Nn);

  // ---- adjacency loss ----
  k_loss1<<<B, 128, 0, stream>>>(out1, dadj, lossA);
  k_loss2<<<B, blk, 0, stream>>>(dadj, sadj, lossB);
  k_loss_final<<<1, blk, 0, stream>>>(lossA, lossB, B, lossOut);

  // ---- dynamic GCN ----
  k_gemm_bf16<<<dim3(C / 64, (int)(RN / 128), 1), blk, 0, stream>>>(
      xTb, WdyT, bufA, (int)RN, C, C, C, C, C, 0, 0, 0);
  k_graph_dyn<<<dim3(C / 256, Nn, B), blk, 0, stream>>>(dadj, Dd, bufA, bufB);
  k_colsum_partial<<<dim3(C / 256, 80), blk, 0, stream>>>(bufB, (int)RN, C, 256, ps, pq);
  k_colsum_final<<<C / 256, blk, 0, stream>>>(ps, pq, C, 80, 1.0f / (float)RN, mu1, var1);
  k_bn_leaky_out<<<(int)((tX + 255) / 256), blk, 0, stream>>>(bufB, mu1, var1, g1, b1, fout, B, C, Nn);
}
